// SelfAttention_13795434954889
// MI455X (gfx1250) — matrix operand order repaired and checked by reading, not compiled, not run on hardware
//
#include <hip/hip_runtime.h>
#include <hip/hip_bf16.h>

typedef __attribute__((ext_vector_type(16))) __bf16 v16bf;
typedef __attribute__((ext_vector_type(8)))  float  v8f;

#define B_   4
#define S_   4096
#define D_   1024
#define QT   16
#define KT   32
#define SCALE 0.03125f   // 1/sqrt(1024)

union BF16Vec {
    v16bf  v;
    uint4  u[2];
    __bf16 e[16];
};

__device__ inline void cvt8(BF16Vec& d, int base, float4 a, float4 b) {
    d.e[base + 0] = (__bf16)a.x; d.e[base + 1] = (__bf16)a.y;
    d.e[base + 2] = (__bf16)a.z; d.e[base + 3] = (__bf16)a.w;
    d.e[base + 4] = (__bf16)b.x; d.e[base + 5] = (__bf16)b.y;
    d.e[base + 6] = (__bf16)b.z; d.e[base + 7] = (__bf16)b.w;
}

// Async global -> LDS copy of 16 bytes (ASYNCcnt tracked, GVS addressing mode)
__device__ inline void async_b128(unsigned ldsOff, unsigned vOff,
                                  unsigned long long sBase) {
    asm volatile("global_load_async_to_lds_b128 %0, %1, %2 offset:0"
                 :: "v"(ldsOff), "v"(vOff), "s"(sBase) : "memory");
}
__device__ inline void wait_async0() {
    asm volatile("s_wait_asynccnt 0x0" ::: "memory");
}

// ---------------------------------------------------------------------------
// Weight convert: Wt[out][in] bf16 = (bf16) W[in][out]  (64x64 LDS transpose)
// ---------------------------------------------------------------------------
__global__ __launch_bounds__(256) void wcvt_kernel(
    const float* __restrict__ W, __bf16* __restrict__ Wt)
{
    __shared__ float t[64][65];
    const int tid = threadIdx.x;
    const int k0  = blockIdx.x * 64;
    const int n0  = blockIdx.y * 64;
    #pragma unroll
    for (int i = 0; i < 16; ++i) {
        int e = tid + i * 256, r = e >> 6, c = e & 63;
        t[r][c] = W[(size_t)(k0 + r) * D_ + n0 + c];
    }
    __syncthreads();
    #pragma unroll
    for (int i = 0; i < 16; ++i) {
        int e = tid + i * 256, r = e >> 6, c = e & 63;
        Wt[(size_t)(n0 + r) * D_ + k0 + c] = (__bf16)t[c][r];
    }
}

// ---------------------------------------------------------------------------
// Projection: Y = X @ W + bias, using pre-transposed bf16 Wt[out][in].
// Block: 256 thr = 8 waves; tile M=16, N=128. No LDS, no barriers.
// ---------------------------------------------------------------------------
__global__ __launch_bounds__(256) void proj_kernel(
    const float*  __restrict__ X,     // [16384, 1024] f32
    const __bf16* __restrict__ Wt,    // [1024 out][1024 in] bf16
    const float*  __restrict__ bias,  // [1024]
    __bf16* __restrict__ Y,           // row-major [16384,1024] or [B][D][S]
    int transposeV)
{
    const int tid  = threadIdx.x;
    const int lane = tid & 31;
    const int wv   = tid >> 5;
    const int hh   = lane >> 4;
    const int l16  = lane & 15;

    const int m0 = blockIdx.x * QT;
    const int n0 = blockIdx.y * 128 + wv * 16;

    v8f acc = {};
    const float*  xrow = X  + (size_t)(m0 + l16) * D_;
    const __bf16* wrow = Wt + (size_t)(n0 + l16) * D_;

    for (int kb = 0; kb < D_; kb += KT) {
        // A operand (16x32 bf16): fp32 X -> bf16 on the fly
        BF16Vec a;
        float4 f0 = *(const float4*)(xrow + kb + hh * 8);
        float4 f1 = *(const float4*)(xrow + kb + hh * 8 + 4);
        float4 f2 = *(const float4*)(xrow + kb + 16 + hh * 8);
        float4 f3 = *(const float4*)(xrow + kb + 16 + hh * 8 + 4);
        cvt8(a, 0, f0, f1);
        cvt8(a, 8, f2, f3);

        // B operand (32x16 bf16): table k order from transposed weights (L2)
        BF16Vec b;
        b.u[0] = *(const uint4*)(wrow + kb + hh * 8);
        b.u[1] = *(const uint4*)(wrow + kb + 16 + hh * 8);

        acc = __builtin_amdgcn_wmma_f32_16x16x32_bf16(
                  false, a.v, false, b.v, (short)0, acc, false, false);
    }

    const float bb = bias[n0 + l16];

    if (!transposeV) {
        #pragma unroll
        for (int r = 0; r < 8; ++r) {
            int g = m0 + r + 8 * hh;
            Y[(size_t)g * D_ + n0 + l16] = (__bf16)(acc[r] + bb);
        }
    } else {
        BF16Vec p;
        #pragma unroll
        for (int r = 0; r < 8; ++r) p.e[r] = (__bf16)(acc[r] + bb);
        const int bidx = m0 >> 12;
        const int s    = (m0 & (S_ - 1)) + 8 * hh;
        __bf16* dst = Y + ((size_t)bidx * D_ + n0 + l16) * S_ + s;
        *(uint4*)dst = p.u[0];
    }
}

// ---------------------------------------------------------------------------
// Flash attention: one block per (batch, 16-row q tile), 8 waves each owning a
// 128-wide D slice. Double-buffered async K tiles; V streamed from L2.
// ---------------------------------------------------------------------------
__global__ __launch_bounds__(256) void attn_kernel(
    const __bf16* __restrict__ Qg,   // [B*S, D] bf16
    const __bf16* __restrict__ Kg,   // [B*S, D] bf16
    const __bf16* __restrict__ Vt,   // [B, D, S] bf16
    float* __restrict__ Out)         // [B, S, D] f32
{
    extern __shared__ char smem_raw[];
    __bf16* Qs   = (__bf16*)smem_raw;            // [QT][D]      32 KB
    __bf16* Ks0  = Qs + QT * D_;                 // [KT][D]      64 KB
    __bf16* Ks1  = Ks0 + KT * D_;                // [KT][D]      64 KB
    float*  Sred = (float*)(Ks1 + KT * D_);      // [QT][KT]      2 KB
    __bf16* Pb   = (__bf16*)(Sred + QT * KT);    // [QT][KT]      1 KB
    float*  mrow = (float*)(Pb + QT * KT);
    float*  lrow = mrow + QT;
    float*  arow = lrow + QT;

    const int tid  = threadIdx.x;
    const int lane = tid & 31;
    const int wv   = tid >> 5;
    const int hh   = lane >> 4;
    const int l16  = lane & 15;

    const int b  = blockIdx.x >> 8;
    const int q0 = (blockIdx.x & 255) * QT;

    const __bf16* Qb  = Qg + ((size_t)b * S_ + q0) * D_;
    const __bf16* Kb  = Kg + (size_t)b * S_ * D_;
    const __bf16* Vtb = Vt + (size_t)b * D_ * S_;

    const unsigned long long qb64 = (unsigned long long)(uintptr_t)Qb;
    const unsigned long long kb64 = (unsigned long long)(uintptr_t)Kb;
    const unsigned qsOff  = (unsigned)(uintptr_t)Qs;   // LDS byte offsets
    const unsigned ksOff0 = (unsigned)(uintptr_t)Ks0;
    const unsigned ksOff1 = (unsigned)(uintptr_t)Ks1;

    // Prologue: async-load Q tile (32 KB) and K tile 0 (64 KB)
    #pragma unroll
    for (int i = 0; i < 8; ++i)
        async_b128(qsOff + tid * 16 + i * 4096, tid * 16 + i * 4096, qb64);
    #pragma unroll
    for (int i = 0; i < 16; ++i)
        async_b128(ksOff0 + tid * 16 + i * 4096, tid * 16 + i * 4096, kb64);

    if (tid < QT) { mrow[tid] = -3.0e38f; lrow[tid] = 0.f; }

    v8f o[8];
    #pragma unroll
    for (int j = 0; j < 8; ++j) o[j] = (v8f){};

    const int dslice = wv * 128;

    wait_async0();
    __syncthreads();

    int cur = 0;
    for (int kb = 0; kb < S_ / KT; ++kb) {
        const int k0 = kb * KT;
        const bool last = (kb == S_ / KT - 1);

        // Kick off async load of the NEXT K tile into the other buffer
        if (!last) {
            const unsigned nOff  = cur ? ksOff0 : ksOff1;
            const unsigned vbase = (unsigned)(k0 + KT) * (D_ * 2);
            #pragma unroll
            for (int i = 0; i < 16; ++i)
                async_b128(nOff + tid * 16 + i * 4096,
                           vbase + tid * 16 + i * 4096, kb64);
        }
        const __bf16* Kcur = cur ? Ks1 : Ks0;

        Sred[tid]       = 0.f;
        Sred[tid + 256] = 0.f;
        __syncthreads();

        // Partial S = Q . K^T over this wave's 128-wide D slice
        v8f s0 = {}, s1 = {};
        #pragma unroll
        for (int dk = 0; dk < 4; ++dk) {
            const int kd = dslice + dk * 32;
            BF16Vec a, b0, b1;
            const __bf16* qa = Qs + l16 * D_ + kd;
            a.u[0] = *(const uint4*)(qa + hh * 8);
            a.u[1] = *(const uint4*)(qa + 16 + hh * 8);
            const __bf16* kc0 = Kcur + l16 * D_ + kd + hh * 8;
            b0.u[0] = *(const uint4*)kc0;
            b0.u[1] = *(const uint4*)(kc0 + 16);
            const __bf16* kc1 = Kcur + (l16 + 16) * D_ + kd + hh * 8;
            b1.u[0] = *(const uint4*)kc1;
            b1.u[1] = *(const uint4*)(kc1 + 16);
            s0 = __builtin_amdgcn_wmma_f32_16x16x32_bf16(
                     false, a.v, false, b0.v, (short)0, s0, false, false);
            s1 = __builtin_amdgcn_wmma_f32_16x16x32_bf16(
                     false, a.v, false, b1.v, (short)0, s1, false, false);
        }
        // Cross-wave reduction of partial scores (ds_add_f32)
        #pragma unroll
        for (int r = 0; r < 8; ++r) {
            atomicAdd(&Sred[(r + 8 * hh) * KT + l16],      s0[r]);
            atomicAdd(&Sred[(r + 8 * hh) * KT + l16 + 16], s1[r]);
        }
        __syncthreads();

        // Online-softmax row stats (16 rows)
        if (tid < QT) {
            const float* srow = Sred + tid * KT;
            float mx = -3.0e38f;
            #pragma unroll
            for (int j = 0; j < KT; ++j) mx = fmaxf(mx, srow[j] * SCALE);
            const float mold  = mrow[tid];
            const float mnew  = fmaxf(mold, mx);
            const float alpha = __expf(mold - mnew);
            float sum = 0.f;
            #pragma unroll
            for (int j = 0; j < KT; ++j) sum += __expf(srow[j] * SCALE - mnew);
            mrow[tid] = mnew;
            lrow[tid] = lrow[tid] * alpha + sum;
            arow[tid] = alpha;
        }
        __syncthreads();

        // P = exp(S*scale - m) as bf16
        #pragma unroll
        for (int i = 0; i < 2; ++i) {
            int e = tid + i * 256;
            int r = e >> 5;
            Pb[e] = (__bf16)__expf(Sred[e] * SCALE - mrow[r]);
        }
        __syncthreads();

        // Rescale O by alpha, then O += P @ V (B operands streamed from L2)
        float ar[8];
        #pragma unroll
        for (int r = 0; r < 8; ++r) ar[r] = arow[r + 8 * hh];
        #pragma unroll
        for (int j = 0; j < 8; ++j)
            #pragma unroll
            for (int r = 0; r < 8; ++r) o[j][r] *= ar[r];

        BF16Vec pa;
        const __bf16* pr = Pb + l16 * KT;
        pa.u[0] = *(const uint4*)(pr + hh * 8);
        pa.u[1] = *(const uint4*)(pr + 16 + hh * 8);

        #pragma unroll
        for (int j = 0; j < 8; ++j) {
            const int dcol = dslice + j * 16 + l16;
            const __bf16* vc = Vtb + (size_t)dcol * S_ + k0 + hh * 8;
            BF16Vec vb;
            vb.u[0] = *(const uint4*)vc;
            vb.u[1] = *(const uint4*)(vc + 16);
            if (!last) __builtin_prefetch(vc + KT, 0, 0);  // next k-step V
            o[j] = __builtin_amdgcn_wmma_f32_16x16x32_bf16(
                       false, pa.v, false, vb.v, (short)0, o[j], false, false);
        }

        // Next K tile must be fully in LDS before anyone reads it
        wait_async0();
        __syncthreads();
        cur ^= 1;
    }

    // Epilogue: normalize by l and store fp32
    float li[8];
    #pragma unroll
    for (int r = 0; r < 8; ++r) li[r] = 1.0f / lrow[r + 8 * hh];
    float* outb = Out + ((size_t)b * S_ + q0) * D_;
    #pragma unroll
    for (int j = 0; j < 8; ++j)
        #pragma unroll
        for (int r = 0; r < 8; ++r)
            outb[(size_t)(r + 8 * hh) * D_ + dslice + j * 16 + l16] = o[j][r] * li[r];
}

// ---------------------------------------------------------------------------
extern "C" void kernel_launch(void* const* d_in, const int* in_sizes, int n_in,
                              void* d_out, int out_size, void* d_ws, size_t ws_size,
                              hipStream_t stream) {
    const float* x  = (const float*)d_in[0];
    const float* Wq = (const float*)d_in[1];
    const float* bq = (const float*)d_in[2];
    const float* Wk = (const float*)d_in[3];
    const float* bk = (const float*)d_in[4];
    const float* Wv = (const float*)d_in[5];
    const float* bv = (const float*)d_in[6];
    float* out = (float*)d_out;

    const size_t NE = (size_t)B_ * S_ * D_;   // 16,777,216
    const size_t WE = (size_t)D_ * D_;        // 1,048,576
    __bf16* Qb  = (__bf16*)d_ws;              // [B*S, D]
    __bf16* Kb  = Qb + NE;                    // [B*S, D]
    __bf16* Vtb = Kb + NE;                    // [B, D, S]
    __bf16* WtQ = Vtb + NE;                   // [D, D] transposed bf16
    __bf16* WtK = WtQ + WE;
    __bf16* WtV = WtK + WE;

    dim3 wgrid(D_ / 64, D_ / 64);             // (16,16)
    wcvt_kernel<<<wgrid, 256, 0, stream>>>(Wq, WtQ);
    wcvt_kernel<<<wgrid, 256, 0, stream>>>(Wk, WtK);
    wcvt_kernel<<<wgrid, 256, 0, stream>>>(Wv, WtV);

    dim3 pgrid(B_ * S_ / QT, D_ / 128);       // (1024, 8)
    proj_kernel<<<pgrid, 256, 0, stream>>>(x, WtQ, bq, Qb, 0);
    proj_kernel<<<pgrid, 256, 0, stream>>>(x, WtK, bk, Kb, 0);
    proj_kernel<<<pgrid, 256, 0, stream>>>(x, WtV, bv, Vtb, 1);

    const size_t smem = (size_t)(QT * D_ + 2 * KT * D_) * sizeof(__bf16)
                      + (size_t)QT * KT * sizeof(float)
                      + (size_t)QT * KT * sizeof(__bf16)
                      + 3u * QT * sizeof(float);
    attn_kernel<<<dim3(B_ * S_ / QT), 256, smem, stream>>>(Qb, Kb, Vtb, out);
}
